// PhotonicSelfAttention_24369644437804
// MI455X (gfx1250) — matrix operand order repaired and checked by reading, not compiled, not run on hardware
//
#include <hip/hip_runtime.h>
#include <hip/hip_bf16.h>

// ---------------------------------------------------------------------------
// PhotonicSelfAttention for MI455X (gfx1250, wave32, WMMA).
//   B=2, S=2048, D=1024, H=16, HD=64. fp32 I/O, f16 WMMA with f32 accumulate.
// Workspace layout (needs 48 MB):
//   [0,8M)   X16   f16 [4096,1024]
//   [8,16M)  Q16   f16 [4096,1024]
//   [16,24M) K16   f16 [4096,1024]
//   [24,32M) Vt16  f16 [B,H,64,2048]   (head-transposed V)
//   [32,40M) A16   f16 [4096,1024]     (attention output)
//   [40,48M) WqT,WkT,WvT,WoT  f16 [1024,1024] each (transposed weights)
// ---------------------------------------------------------------------------

typedef _Float16 half_t;
typedef __attribute__((ext_vector_type(16))) _Float16 v16h;
typedef __attribute__((ext_vector_type(8)))  _Float16 v8h;
typedef __attribute__((ext_vector_type(4)))  _Float16 v4h;
typedef __attribute__((ext_vector_type(8)))  float    v8f;

#define CB  2
#define CS  2048
#define CD  1024
#define CH  16
#define CHD 64
#define CM  (CB * CS)          // 4096 rows of X/Q/K/V/attn
#define PROW 88                // LDS P-tile row stride in halfs (64 + 24 pad)

__device__ __forceinline__ v16h cat8(v8h lo, v8h hi) {
  return __builtin_shufflevector(lo, hi, 0,1,2,3,4,5,6,7,8,9,10,11,12,13,14,15);
}

__device__ __forceinline__ v8f wmma_f16(v16h a, v16h b, v8f c) {
  // D = A(16x32 f16) x B(32x16 f16) + C(16x16 f32)
  return __builtin_amdgcn_wmma_f32_16x16x32_f16(false, a, false, b, (short)0, c,
                                                false, false);
}

// max-reduce across the 16-lane half-wave that shares a C-fragment row set
__device__ __forceinline__ float rmax16(float v) {
#pragma unroll
  for (int m = 8; m >= 1; m >>= 1) v = fmaxf(v, __shfl_xor(v, m, 32));
  return v;
}

// load a 16-bit A/B fragment as two 16B loads
__device__ __forceinline__ v16h fragAB(const half_t* p, int hiOff) {
  v8h lo = *reinterpret_cast<const v8h*>(p);
  v8h hi = *reinterpret_cast<const v8h*>(p + hiOff);
  return cat8(lo, hi);
}

// ---------------------------------------------------------------------------
// fp32 -> f16 elementwise (float4 granularity)
// ---------------------------------------------------------------------------
__global__ __launch_bounds__(256) void cvt_f32_f16(const float* __restrict__ in,
                                                   half_t* __restrict__ out,
                                                   int n4) {
  int i = blockIdx.x * 256 + threadIdx.x;
  if (i >= n4) return;
  float4 v = reinterpret_cast<const float4*>(in)[i];
  v4h o;
  o[0] = (half_t)v.x; o[1] = (half_t)v.y; o[2] = (half_t)v.z; o[3] = (half_t)v.w;
  reinterpret_cast<v4h*>(out)[i] = o;
}

// ---------------------------------------------------------------------------
// W [K=1024, N=1024] fp32 row-major  ->  WT [N, K] f16 row-major (LDS tiled)
// ---------------------------------------------------------------------------
__global__ __launch_bounds__(256) void wtrans_f16(const float* __restrict__ W,
                                                  half_t* __restrict__ WT) {
  __shared__ float t[16][17];
  int n0 = blockIdx.x * 16, k0 = blockIdx.y * 16;
  int r = threadIdx.x >> 4, c = threadIdx.x & 15;
  t[r][c] = W[(size_t)(k0 + r) * CD + n0 + c];
  __syncthreads();
  WT[(size_t)(n0 + r) * CD + k0 + c] = (half_t)t[c][r];
}

// ---------------------------------------------------------------------------
// C[M=4096, N=1024] = A16[M,1024] @ WT^T + bias
//   MODE 0: f16 store, row-major [M,N]
//   MODE 1: f16 store, head-transposed [B,H,HD,S]   (for V)
//   MODE 2: f32 store, row-major [M,N]              (final projection)
// One wave computes a 32x64 tile (8 accumulators, 8 WMMA / 12 b128 per iter).
// Grid = 256 blocks x 256 threads (2048 waves).
// ---------------------------------------------------------------------------
template <int MODE>
__global__ __launch_bounds__(256) void gemm_wmma(const half_t* __restrict__ A,
                                                 const half_t* __restrict__ WT,
                                                 const float* __restrict__ bias,
                                                 void* __restrict__ Cv) {
  const int tid = threadIdx.x;
  const int lane = tid & 31, lane16 = lane & 15, g = lane >> 4;
  const int wid = (blockIdx.x * 256 + tid) >> 5;   // 0..2047
  const int mt  = wid & 127;                        // 128 M tiles of 32 rows
  const int n64 = wid >> 7;                         // 16 N groups of 64
  const int kbA = g ? 16 : 0;                       // A-frag K base (16-bit layout)
  const int kbB = g ? 16 : 0;                       // B-frag K base

  const half_t* ap0 = A + (size_t)(mt * 32 + lane16) * CD + kbA;
  const half_t* ap1 = ap0 + (size_t)16 * CD;
  const half_t* bp[4];
#pragma unroll
  for (int nt = 0; nt < 4; ++nt)
    bp[nt] = WT + (size_t)(n64 * 64 + nt * 16 + lane16) * CD + kbB;

  v8f z = {};
  v8f acc[2][4] = {{z, z, z, z}, {z, z, z, z}};

  for (int k0 = 0; k0 < CD; k0 += 32) {
    v16h a0 = fragAB(ap0 + k0, 8);
    v16h a1 = fragAB(ap1 + k0, 8);
#pragma unroll
    for (int nt = 0; nt < 4; ++nt) {
      v16h bb = fragAB(bp[nt] + k0, 8);
      acc[0][nt] = wmma_f16(a0, bb, acc[0][nt]);
      acc[1][nt] = wmma_f16(a1, bb, acc[1][nt]);
    }
  }

#pragma unroll
  for (int nt = 0; nt < 4; ++nt) {
    const int col = n64 * 64 + nt * 16 + lane16;
    const float bv = bias[col];
#pragma unroll
    for (int mi = 0; mi < 2; ++mi) {
#pragma unroll
      for (int r = 0; r < 8; ++r) {
        const int row = mt * 32 + mi * 16 + g * 8 + r;
        const float val = acc[mi][nt][r] + bv;
        if (MODE == 0) {
          reinterpret_cast<half_t*>(Cv)[(size_t)row * CD + col] = (half_t)val;
        } else if (MODE == 1) {
          const int b = row >> 11, s = row & (CS - 1);
          const int h = col >> 6,  d = col & (CHD - 1);
          reinterpret_cast<half_t*>(Cv)[(((size_t)(b * CH + h) * CHD + d) << 11) + s] =
              (half_t)val;
        } else {
          reinterpret_cast<float*>(Cv)[(size_t)row * CD + col] = val;
        }
      }
    }
  }
}

// ---------------------------------------------------------------------------
// Flash attention: one wave per 32-query tile of one (b,h), 64 keys per step.
//   Q16/K16: f16 [4096,1024], Vt: f16 [B,H,64,S], out A16: f16 [4096,1024]
// Per step: 16 WMMA (Q.K^T) + 4 WMMA row-sums + 16 WMMA (P.V), with each K and
// V fragment (b128 pair) reused by both query sub-tiles (~0.9 loads/WMMA).
// Score accumulator jt holds keys {j0 + lane16*4 + jt}: each lane writes its 4
// probabilities as one packed ds_store_b64, while the LDS P tile stays in
// absolute-key order so P A-fragments re-load as contiguous b128s.
// Interleaved qi-WMMAs are independent, hiding WMMA->WMMA RAW hazard NOPs.
// ---------------------------------------------------------------------------
__global__ __launch_bounds__(256) void attn_fwd(const half_t* __restrict__ Q,
                                                const half_t* __restrict__ Kg,
                                                const half_t* __restrict__ Vt,
                                                half_t* __restrict__ O) {
  __shared__ __align__(16) half_t Pb[8][2][16 * PROW];  // per-wave P tiles

  const int tid = threadIdx.x;
  const int lane = tid & 31, lane16 = lane & 15, g = lane >> 4, w = tid >> 5;
  const int wid = blockIdx.x * 8 + w;              // 0..2047
  const int bh = wid >> 6;                         // 64 q-tiles of 32 per (b,h)
  const int qt = wid & 63;
  const int b = bh >> 4, h = bh & 15;
  const int q0 = qt * 32;
  const int kbA = g ? 16 : 0;
  const int kbB = g ? 16 : 0;

  // Q A-fragments [qi][d-chunk], pre-scaled by 1/sqrt(64)
  v16h qa[2][2];
#pragma unroll
  for (int qi = 0; qi < 2; ++qi) {
    const half_t* qrow =
        Q + (size_t)(b * CS + q0 + qi * 16 + lane16) * CD + h * CHD;
#pragma unroll
    for (int c = 0; c < 2; ++c) {
      v16h t = fragAB(qrow + c * 32 + kbA, 8);
#pragma unroll
      for (int i = 0; i < 16; ++i) t[i] *= (_Float16)0.125f;
      qa[qi][c] = t;
    }
  }
  // all-ones B fragment for WMMA row-sums
  v16h ones;
#pragma unroll
  for (int i = 0; i < 16; ++i) ones[i] = (_Float16)1.0f;

  v8f z = {};
  v8f accO[2][4] = {{z, z, z, z}, {z, z, z, z}};
  float mrow[2][8], lrow[2][8];
#pragma unroll
  for (int qi = 0; qi < 2; ++qi)
#pragma unroll
    for (int r = 0; r < 8; ++r) { mrow[qi][r] = -1e30f; lrow[qi][r] = 0.f; }

  const half_t* kbase = Kg + (size_t)(b * CS) * CD + h * CHD + kbB;
  const half_t* vbase = Vt + (size_t)bh * CHD * CS + kbB;

  for (int j0 = 0; j0 < CS; j0 += 64) {
    // ---- scores for 64 keys: acc jt holds keys j0 + lane16*4 + jt ----
    v8f s[2][4] = {{z, z, z, z}, {z, z, z, z}};
#pragma unroll
    for (int jt = 0; jt < 4; ++jt) {
      const half_t* krow = kbase + (size_t)(j0 + lane16 * 4 + jt) * CD;
      v16h kf0 = fragAB(krow, 8);
      v16h kf1 = fragAB(krow + 32, 8);
#pragma unroll
      for (int qi = 0; qi < 2; ++qi) {
        s[qi][jt] = wmma_f16(qa[qi][0], kf0, s[qi][jt]);
        s[qi][jt] = wmma_f16(qa[qi][1], kf1, s[qi][jt]);
      }
    }
    // ---- online softmax: row-max (shuffle), P -> LDS (packed b64) ----
#pragma unroll
    for (int qi = 0; qi < 2; ++qi) {
      half_t* pb = Pb[w][qi];
#pragma unroll
      for (int r = 0; r < 8; ++r) {
        const float rm = rmax16(fmaxf(fmaxf(s[qi][0][r], s[qi][1][r]),
                                      fmaxf(s[qi][2][r], s[qi][3][r])));
        const float mnew = fmaxf(mrow[qi][r], rm);
        const float corr = __expf(mrow[qi][r] - mnew);
        v4h pv;
#pragma unroll
        for (int jt = 0; jt < 4; ++jt)
          pv[jt] = (half_t)__expf(s[qi][jt][r] - mnew);
        lrow[qi][r] *= corr;
        mrow[qi][r] = mnew;
#pragma unroll
        for (int dt = 0; dt < 4; ++dt) accO[qi][dt][r] *= corr;
        const int m = g * 8 + r;                      // C-layout row -> P row
        *reinterpret_cast<v4h*>(pb + m * PROW + lane16 * 4) = pv;
      }
    }
    // in-wave LDS ordering: stores complete before b128 re-reads
    __asm__ volatile("s_wait_dscnt 0" ::: "memory");
    // ---- P as A-fragments [qi][key-chunk]; row-sums via WMMA ----
    v16h pA[2][2];
#pragma unroll
    for (int qi = 0; qi < 2; ++qi) {
      const half_t* prow = Pb[w][qi] + lane16 * PROW + kbA;
      pA[qi][0] = fragAB(prow, 8);         // keys j0 .. j0+31
      pA[qi][1] = fragAB(prow + 32, 8);    // keys j0+32 .. j0+63
      v8f sums = wmma_f16(pA[qi][0], ones, z);
      sums = wmma_f16(pA[qi][1], ones, sums);
#pragma unroll
      for (int r = 0; r < 8; ++r) lrow[qi][r] += sums[r];
    }
    // ---- O += P.V  (V^T fragments shared by both query sub-tiles) ----
#pragma unroll
    for (int dt = 0; dt < 4; ++dt) {
      const half_t* vrow = vbase + (size_t)(dt * 16 + lane16) * CS + j0;
      v16h vf0 = fragAB(vrow, 8);
      v16h vf1 = fragAB(vrow + 32, 8);
#pragma unroll
      for (int qi = 0; qi < 2; ++qi) {
        accO[qi][dt] = wmma_f16(pA[qi][0], vf0, accO[qi][dt]);
        accO[qi][dt] = wmma_f16(pA[qi][1], vf1, accO[qi][dt]);
      }
    }
  }

  // ---- epilogue: O /= l, store f16 row-major [4096,1024] ----
#pragma unroll
  for (int qi = 0; qi < 2; ++qi)
#pragma unroll
    for (int r = 0; r < 8; ++r) {
      const float inv = 1.0f / lrow[qi][r];
      const size_t row = (size_t)(b * CS + q0 + qi * 16 + g * 8 + r);
#pragma unroll
      for (int dt = 0; dt < 4; ++dt)
        O[row * CD + h * CHD + dt * 16 + lane16] =
            (half_t)(accO[qi][dt][r] * inv);
    }
}

// ---------------------------------------------------------------------------
extern "C" void kernel_launch(void* const* d_in, const int* in_sizes, int n_in,
                              void* d_out, int out_size, void* d_ws, size_t ws_size,
                              hipStream_t stream) {
  (void)in_sizes; (void)n_in; (void)out_size; (void)ws_size;
  const float* x  = (const float*)d_in[0];
  const float* Wq = (const float*)d_in[1];
  const float* bq = (const float*)d_in[2];
  const float* Wk = (const float*)d_in[3];
  const float* bk = (const float*)d_in[4];
  const float* Wv = (const float*)d_in[5];
  const float* bv = (const float*)d_in[6];
  const float* Wo = (const float*)d_in[7];
  const float* bo = (const float*)d_in[8];

  char* ws = (char*)d_ws;
  const size_t MB = 1024ull * 1024ull;
  half_t* X16 = (half_t*)(ws + 0 * MB);
  half_t* Q16 = (half_t*)(ws + 8 * MB);
  half_t* K16 = (half_t*)(ws + 16 * MB);
  half_t* Vt  = (half_t*)(ws + 24 * MB);
  half_t* A16 = (half_t*)(ws + 32 * MB);
  half_t* WqT = (half_t*)(ws + 40 * MB);
  half_t* WkT = (half_t*)(ws + 42 * MB);
  half_t* WvT = (half_t*)(ws + 44 * MB);
  half_t* WoT = (half_t*)(ws + 46 * MB);

  // 1) activations fp32 -> f16
  cvt_f32_f16<<<(CM * CD / 4 + 255) / 256, 256, 0, stream>>>(x, X16, CM * CD / 4);
  // 2) weights fp32 [K,N] -> f16 [N,K]
  dim3 tg(CD / 16, CD / 16);
  wtrans_f16<<<tg, 256, 0, stream>>>(Wq, WqT);
  wtrans_f16<<<tg, 256, 0, stream>>>(Wk, WkT);
  wtrans_f16<<<tg, 256, 0, stream>>>(Wv, WvT);
  wtrans_f16<<<tg, 256, 0, stream>>>(Wo, WoT);
  // 3) QKV projections (V stored head-transposed [B,H,HD,S])
  gemm_wmma<0><<<256, 256, 0, stream>>>(X16, WqT, bq, (void*)Q16);
  gemm_wmma<0><<<256, 256, 0, stream>>>(X16, WkT, bk, (void*)K16);
  gemm_wmma<1><<<256, 256, 0, stream>>>(X16, WvT, bv, (void*)Vt);
  // 4) flash attention (32-query tiles)
  attn_fwd<<<256, 256, 0, stream>>>(Q16, K16, Vt, A16);
  // 5) output projection -> fp32 d_out
  gemm_wmma<2><<<256, 256, 0, stream>>>(A16, WoT, bo, d_out);
}
